// SelfAttention_46007689675365
// MI455X (gfx1250) — matrix operand order repaired and checked by reading, compiled, not run on hardware
//
#include <hip/hip_runtime.h>
#include <hip/hip_bf16.h>

typedef __attribute__((ext_vector_type(16))) _Float16 v16h;
typedef __attribute__((ext_vector_type(8)))  float    v8f;

#define EMBED 1024
#define HEADS 16
#define HDIM  64
#define NB    4
#define SEQ   2048

static __device__ __forceinline__ v8f wmma_f16(v16h a, v16h b, v8f c) {
  // D = A(16x32 f16) * B(32x16 f16) + C(16x16 f32)
  return __builtin_amdgcn_wmma_f32_16x16x32_f16(
      /*neg_a=*/false, a, /*neg_b=*/false, b,
      /*c_mod=*/(short)0, c, /*reuse_a=*/false, /*reuse_b=*/false);
}

// --------------------------- async LDS staging -----------------------------
#if __has_builtin(__builtin_amdgcn_global_load_async_to_lds_b128)
#define HAVE_ASYNC_LDS 1
typedef int v4i_vs __attribute__((vector_size(16)));
typedef __attribute__((address_space(1))) v4i_vs gv4i_t;   // global int4
typedef __attribute__((address_space(3))) v4i_vs lv4i_t;   // LDS int4
#else
#define HAVE_ASYNC_LDS 0
#endif

// 16 bytes per calling thread: global -> LDS
static __device__ __forceinline__ void tile_load_16B(const _Float16* g,
                                                     _Float16* l) {
#if HAVE_ASYNC_LDS
  __builtin_amdgcn_global_load_async_to_lds_b128((gv4i_t*)g, (lv4i_t*)l,
                                                 /*offset=*/0, /*cpol=*/0);
#else
  *(float4*)l = *(const float4*)g;
#endif
}

static __device__ __forceinline__ void wait_async_done() {
#if HAVE_ASYNC_LDS
#if __has_builtin(__builtin_amdgcn_s_wait_asynccnt)
  __builtin_amdgcn_s_wait_asynccnt(0);
#else
  asm volatile("s_wait_asynccnt 0x0" ::: "memory");
#endif
#endif
}

// ---------------------------------------------------------------------------
// Kernel 0: Wo f32 -> f16 (one-shot; Wo is 4MB, L2-resident afterwards).
// ---------------------------------------------------------------------------
__global__ void __launch_bounds__(256)
cvt_wo_kernel(const float* __restrict__ Wo, _Float16* __restrict__ woh) {
  const int idx = (blockIdx.x * 256 + threadIdx.x) * 4;  // < 1024*1024
  const float4 f = *(const float4*)(Wo + idx);
  _Float16 h[4] = {(_Float16)f.x, (_Float16)f.y, (_Float16)f.z, (_Float16)f.w};
  *(ulong1*)(woh + idx) = *(ulong1*)h;
}

// ---------------------------------------------------------------------------
// Kernel 1: QKV head projections.  Each 16-row strip == ONE token (n,s) with
// the 16 heads as rows (HEADS==16):  strip = n*SEQ + s,  head = row-in-tile.
// Output f16 in [n][h][s][d] layout for attention.
// ---------------------------------------------------------------------------
__global__ void __launch_bounds__(256)
proj_qkv_kernel(const float* __restrict__ q, const float* __restrict__ k,
                const float* __restrict__ v, const float* __restrict__ Wq,
                const float* __restrict__ Wk, const float* __restrict__ Wv,
                _Float16* __restrict__ qh, _Float16* __restrict__ kh,
                _Float16* __restrict__ vh) {
  const int wave = threadIdx.x >> 5;
  const int l    = threadIdx.x & 31;
  const int m    = l & 15;
  const int hi   = l >> 4;
  const int nc   = l & 15;
  const int strip = blockIdx.x * 8 + wave;        // token id = n*SEQ + s
  const int nb = strip >> 11;
  const int ss = strip & (SEQ - 1);

  const float* srcs[3] = {q, k, v};
  const float* Ws[3]   = {Wq, Wk, Wv};
  _Float16*    dsts[3] = {qh, kh, vh};

  const size_t dbase = ((size_t)nb * HEADS * SEQ + ss) * HDIM;

#pragma unroll
  for (int p = 0; p < 3; ++p) {
    const float* X = srcs[p] + (size_t)strip * EMBED;   // 16 heads x 64
    const float* W = Ws[p];
    v16h a0, a1;
#pragma unroll
    for (int e = 0; e < 16; ++e) {
      int kk = e + 16 * hi;
      a0[e] = (_Float16)X[m * HDIM + kk];
      a1[e] = (_Float16)X[m * HDIM + 32 + kk];
    }
#pragma unroll
    for (int nt = 0; nt < 4; ++nt) {
      v16h b0, b1;   // B[kd][n] = W[nt*16+n][kc*32+kd]
#pragma unroll
      for (int e = 0; e < 16; ++e) {
        int kd = e + 16 * hi;
        b0[e] = (_Float16)W[(nt * 16 + nc) * HDIM + kd];
        b1[e] = (_Float16)W[(nt * 16 + nc) * HDIM + 32 + kd];
      }
      v8f c = {};
      c = wmma_f16(a0, b0, c);
      c = wmma_f16(a1, b1, c);
      // convert all 8 first (distinct regs), then store back-to-back
      _Float16 hv[8];
#pragma unroll
      for (int r = 0; r < 8; ++r) hv[r] = (_Float16)c[r];
      _Float16* dp = dsts[p] + dbase + (size_t)(8 * hi) * (SEQ * HDIM)
                     + nt * 16 + nc;
#pragma unroll
      for (int r = 0; r < 8; ++r)
        dp[(size_t)r * (SEQ * HDIM)] = hv[r];
    }
  }
}

// ---------------------------------------------------------------------------
// Kernel 2: flash attention (online softmax).  All 8 waves of a block share
// one (n,h); K/V 32x64 f16 tiles are double-buffered in LDS via async loads
// (GLOBAL_LOAD_ASYNC_TO_LDS_B128, ASYNCcnt) so tile kt+1 streams while tile
// kt computes.  Score tile staged through LDS (C-layout -> A-layout).
// grid = N*H*(S/128) = 1024 blocks x 256 thr.
// ---------------------------------------------------------------------------
__global__ void __launch_bounds__(256)
flash_attn_kernel(const _Float16* __restrict__ qh,
                  const _Float16* __restrict__ kh,
                  const _Float16* __restrict__ vh,
                  _Float16* __restrict__ ao) {
  __shared__ _Float16 lds_k[2][32 * HDIM];   // 2 x 4KB
  __shared__ _Float16 lds_v[2][32 * HDIM];   // 2 x 4KB
  __shared__ _Float16 lds_p[8][16 * 32];     // 8KB score staging

  const int tid  = threadIdx.x;
  const int wave = tid >> 5;
  const int l    = tid & 31;
  const int m    = l & 15;
  const int hi   = l >> 4;
  const int nc   = l & 15;

  const int qblocks = SEQ / 128;                       // 16
  const int bid = blockIdx.x;
  const int nb  = bid / (HEADS * qblocks);
  const int rem = bid % (HEADS * qblocks);
  const int hh  = rem / qblocks;
  const int qb  = rem % qblocks;
  const int q0  = qb * 128 + wave * 16;

  const _Float16* Q = qh + (((size_t)nb * HEADS + hh) * SEQ) * HDIM;
  const _Float16* K = kh + (((size_t)nb * HEADS + hh) * SEQ) * HDIM;
  const _Float16* V = vh + (((size_t)nb * HEADS + hh) * SEQ) * HDIM;

  // Q A-fragments (persistent across the K loop)
  v16h aq0, aq1;
#pragma unroll
  for (int e = 0; e < 16; ++e) {
    int kk = e + 16 * hi;
    aq0[e] = Q[(q0 + m) * HDIM + kk];
    aq1[e] = Q[(q0 + m) * HDIM + 32 + kk];
  }

  const float SCALE = 0.03125f;   // 1/sqrt(EMBED)
  float mrow[8], lrow[8];
  v8f acc[4];
  {
    v8f z = {};
#pragma unroll
    for (int dt = 0; dt < 4; ++dt) acc[dt] = z;
#pragma unroll
    for (int r = 0; r < 8; ++r) { mrow[r] = -1e30f; lrow[r] = 0.0f; }
  }

  _Float16* lp = &lds_p[wave][0];
  const int NT = SEQ / 32;   // 64 key tiles

  // preload tile 0 (each thread moves 16B of K and 16B of V)
  tile_load_16B(K + tid * 8, &lds_k[0][tid * 8]);
  tile_load_16B(V + tid * 8, &lds_v[0][tid * 8]);

  for (int kt = 0; kt < NT; ++kt) {
    wait_async_done();        // this wave's async loads for tile kt landed
    __syncthreads();          // => every wave's loads landed

    const int cur = kt & 1;
    if (kt + 1 < NT) {        // stream next tile into the other buffer
      const size_t kn = (size_t)(kt + 1) * 32 * HDIM;
      tile_load_16B(K + kn + tid * 8, &lds_k[cur ^ 1][tid * 8]);
      tile_load_16B(V + kn + tid * 8, &lds_v[cur ^ 1][tid * 8]);
    }

    const _Float16* Kt = &lds_k[cur][0];   // [key][d] 32x64
    const _Float16* Vt = &lds_v[cur][0];

    // B fragments of K^T from LDS: contiguous 16-half runs -> ds_load_b128
    v16h bk00, bk10, bk01, bk11;
#pragma unroll
    for (int e = 0; e < 16; ++e) {
      int kd = e + 16 * hi;
      bk00[e] = Kt[nc * HDIM + kd];
      bk10[e] = Kt[nc * HDIM + 32 + kd];
      bk01[e] = Kt[(16 + nc) * HDIM + kd];
      bk11[e] = Kt[(16 + nc) * HDIM + 32 + kd];
    }
    v8f s0 = {}, s1 = {};
    s0 = wmma_f16(aq0, bk00, s0);
    s0 = wmma_f16(aq1, bk10, s0);
    s1 = wmma_f16(aq0, bk01, s1);
    s1 = wmma_f16(aq1, bk11, s1);

    // online softmax: row stats across the 16-lane half-groups
    float mnew[8], alpha[8];
#pragma unroll
    for (int r = 0; r < 8; ++r) {
      s0[r] *= SCALE; s1[r] *= SCALE;
      float t = fmaxf(s0[r], s1[r]);
#pragma unroll
      for (int off = 1; off < 16; off <<= 1)
        t = fmaxf(t, __shfl_xor(t, off, 32));
      mnew[r]  = fmaxf(mrow[r], t);
      alpha[r] = __expf(mrow[r] - mnew[r]);
      mrow[r]  = mnew[r];
    }
#pragma unroll
    for (int dt = 0; dt < 4; ++dt)
#pragma unroll
      for (int r = 0; r < 8; ++r) acc[dt][r] *= alpha[r];

#pragma unroll
    for (int r = 0; r < 8; ++r) {
      s0[r] = __expf(s0[r] - mnew[r]);
      s1[r] = __expf(s1[r] - mnew[r]);
      float t = s0[r] + s1[r];
#pragma unroll
      for (int off = 1; off < 16; off <<= 1)
        t += __shfl_xor(t, off, 32);
      lrow[r] = lrow[r] * alpha[r] + t;
    }

    // stage P (C-layout) into LDS as 16x32 f16 row-major
#pragma unroll
    for (int r = 0; r < 8; ++r) {
      int mm = r + 8 * hi;
      lp[mm * 32 + nc]      = (_Float16)s0[r];
      lp[mm * 32 + 16 + nc] = (_Float16)s1[r];
    }
    __syncthreads();

    // read back in A layout
    v16h ap;
#pragma unroll
    for (int e = 0; e < 16; ++e) {
      int kk = e + 16 * hi;
      ap[e] = lp[m * 32 + kk];
    }
    // P(16x32) @ V(32x64), V from LDS
#pragma unroll
    for (int dt = 0; dt < 4; ++dt) {
      v16h bv;
#pragma unroll
      for (int e = 0; e < 16; ++e) {
        int kk = e + 16 * hi;
        bv[e] = Vt[kk * HDIM + dt * 16 + nc];
      }
      acc[dt] = wmma_f16(ap, bv, acc[dt]);
    }
    __syncthreads();   // all waves done reading cur buffers + lds_p
  }

  // normalize and emit attention output in [n][s][embed] f16 layout
#pragma unroll
  for (int dt = 0; dt < 4; ++dt) {
    _Float16 hv[8];
#pragma unroll
    for (int r = 0; r < 8; ++r) hv[r] = (_Float16)(acc[dt][r] / lrow[r]);
#pragma unroll
    for (int r = 0; r < 8; ++r) {
      int qi = q0 + r + 8 * hi;
      ao[((size_t)nb * SEQ + qi) * EMBED + hh * HDIM + dt * 16 + nc] = hv[r];
    }
  }
}

// ---------------------------------------------------------------------------
// Kernel 3: output projection  [8192 x 1024] @ Wo^T + bo  (f32 out), Wo in f16.
// grid = 1024 blocks x 256 thr; one wave -> one 16x64 output tile.
// ---------------------------------------------------------------------------
__global__ void __launch_bounds__(256)
out_proj_kernel(const _Float16* __restrict__ ao,
                const _Float16* __restrict__ woh,
                const float* __restrict__ bo, float* __restrict__ out) {
  const int wave = threadIdx.x >> 5;
  const int l    = threadIdx.x & 31;
  const int m    = l & 15;
  const int hi   = l >> 4;
  const int nc   = l & 15;

  const int task = blockIdx.x * 8 + wave;     // 0..8191
  const int r0 = (task >> 4) * 16;            // row strip
  const int c0 = (task & 15) * 64;            // output column group

  v8f acc[4];
  {
    v8f z = {};
#pragma unroll
    for (int nt = 0; nt < 4; ++nt) acc[nt] = z;
  }

  for (int kc = 0; kc < EMBED / 32; ++kc) {
    const int kb = kc * 32;
    v16h a;
#pragma unroll
    for (int e = 0; e < 16; ++e) {
      int kk = e + 16 * hi;
      a[e] = ao[(size_t)(r0 + m) * EMBED + kb + kk];
    }
#pragma unroll
    for (int nt = 0; nt < 4; ++nt) {
      v16h b;   // B[kd][n] = Wo[c0+nt*16+n][kb+kd]  (f16)
#pragma unroll
      for (int e = 0; e < 16; ++e) {
        int kd = e + 16 * hi;
        b[e] = woh[(size_t)(c0 + nt * 16 + nc) * EMBED + kb + kd];
      }
      acc[nt] = wmma_f16(a, b, acc[nt]);
    }
  }

#pragma unroll
  for (int nt = 0; nt < 4; ++nt)
#pragma unroll
    for (int r = 0; r < 8; ++r) {
      int row = r0 + r + 8 * hi;
      int col = c0 + nt * 16 + nc;
      out[(size_t)row * EMBED + col] = acc[nt][r] + bo[col];
    }
}

// ---------------------------------------------------------------------------
extern "C" void kernel_launch(void* const* d_in, const int* in_sizes, int n_in,
                              void* d_out, int out_size, void* d_ws, size_t ws_size,
                              hipStream_t stream) {
  const float* q  = (const float*)d_in[0];
  const float* k  = (const float*)d_in[1];
  const float* v  = (const float*)d_in[2];
  const float* Wq = (const float*)d_in[3];
  const float* Wk = (const float*)d_in[4];
  const float* Wv = (const float*)d_in[5];
  const float* Wo = (const float*)d_in[6];
  const float* bo = (const float*)d_in[7];
  float* out = (float*)d_out;

  // f16 staging: qh/kh/vh in [n][h][s][d], ao in [n][s][embed], woh = Wo f16
  const size_t HALF_PER_BUF = (size_t)NB * HEADS * SEQ * HDIM;  // 8,388,608
  _Float16* ws  = (_Float16*)d_ws;
  _Float16* qh  = ws;
  _Float16* kh  = ws + HALF_PER_BUF;
  _Float16* vh  = ws + 2 * HALF_PER_BUF;
  _Float16* ao  = ws + 3 * HALF_PER_BUF;
  _Float16* woh = ws + 4 * HALF_PER_BUF;                        // 1M halves

  cvt_wo_kernel<<<dim3(1024), dim3(256), 0, stream>>>(Wo, woh);
  proj_qkv_kernel<<<dim3(1024), dim3(256), 0, stream>>>(q, k, v, Wq, Wk, Wv,
                                                        qh, kh, vh);
  flash_attn_kernel<<<dim3(1024), dim3(256), 0, stream>>>(qh, kh, vh, ao);
  out_proj_kernel<<<dim3(1024), dim3(256), 0, stream>>>(ao, woh, bo, out);
}
